// UserGAT_42992622633208
// MI455X (gfx1250) — matrix operand order repaired and checked by reading, not compiled, not run on hardware
//
#include <hip/hip_runtime.h>
#include <hip/hip_bf16.h>

typedef __attribute__((ext_vector_type(16))) _Float16     v16h;
typedef __attribute__((ext_vector_type(8)))  float        v8f;
typedef __attribute__((ext_vector_type(4)))  unsigned int v4u;
typedef __attribute__((ext_vector_type(4)))  int          v4i;
typedef __attribute__((ext_vector_type(8)))  int          v8i;

#define N_NODES  50000
#define DIM      64
#define N_EDGES  800000
#define TOT_E    (N_EDGES + N_NODES)
#define NEG_SLOPE 0.2f
#define KEEP_P    0.7f

union HFrag {
  v16h v;
  unsigned int u[8];
  _Float16 h[16];
};

// ---------------------------------------------------------------------------
// Per-layer scratch init: agg = 0, amax = -inf, denom = 0
// ---------------------------------------------------------------------------
__global__ __launch_bounds__(256) void gat_init(float* __restrict__ agg,
                                                float* __restrict__ amax,
                                                float* __restrict__ denom) {
  int i = blockIdx.x * 256 + threadIdx.x;
  if (i < N_NODES * DIM) agg[i] = 0.0f;
  if (i < N_NODES) { amax[i] = -__builtin_inff(); denom[i] = 0.0f; }
}

// ---------------------------------------------------------------------------
// Fused: h = x @ W (WMMA f16->f32), s_src = h.a_src, s_dst = h.a_dst
// One wave: 16 nodes x 64 features = 4 C-tiles, K=64 in 2 chunks -> 8 WMMAs.
// W is DMA'd into LDS by the Tensor Data Mover (tensor_load_to_lds), then
// transposed/converted to f16 so B fragments are contiguous dwords.
// ---------------------------------------------------------------------------
__global__ __launch_bounds__(256) void gat_gemm_attn(
    const float* __restrict__ x, const float* __restrict__ W,
    const float* __restrict__ a_src, const float* __restrict__ a_dst,
    float* __restrict__ h, float* __restrict__ ssrc, float* __restrict__ sdst) {
  // f32 staging buffer at LDS offset 0 (matches D#.lds_addr = 0), f16 after it
  __shared__ __align__(16) char smem[DIM * DIM * 4 + DIM * DIM * 2];
  float*    Wf32 = (float*)smem;
  _Float16* Wlds = (_Float16*)(smem + DIM * DIM * 4);  // transposed [n][k]

  int tid = threadIdx.x;

#if __has_builtin(__builtin_amdgcn_tensor_load_to_lds)
  // ---- TDM: one wave DMAs the whole 64x64 f32 W tile into LDS ----
  if (tid < 32) {
    unsigned long long ga = (unsigned long long)(const void*)W;
    // D# group0: count=1 | lds_addr=0 | global_addr | type=2
    v4u g0 = { 1u,
               0u,
               (unsigned int)ga,
               ((unsigned int)(ga >> 32) & 0x1FFFFFFu) | (2u << 30) };
    // D# group1: data_size=4B; tensor_dim0 = tile_dim0 = stride = 4096 elems;
    //            tensor_dim1 = tile_dim1 = 1 (one 16KB row)
    v8i g1 = { (int)(2u << 16),          // data_size = 2 (4 bytes)
               (int)(0x1000u << 16),     // tensor_dim0 lo16 = 4096
               (int)(1u << 16),          // tensor_dim0 hi16=0 | tensor_dim1 lo16=1
               (int)(0x1000u << 16),     // tensor_dim1 hi16=0 | tile_dim0=4096
               1,                        // tile_dim1=1 | tile_dim2=0
               4096,                     // tensor_dim0_stride lo32
               0, 0 };
    v4i g2 = { 0, 0, 0, 0 };
    v4i g3 = { 0, 0, 0, 0 };
#if defined(__clang_major__) && __clang_major__ >= 23
    v8i g4 = { 0, 0, 0, 0, 0, 0, 0, 0 };
    __builtin_amdgcn_tensor_load_to_lds(g0, g1, g2, g3, g4, 0);
#else
    __builtin_amdgcn_tensor_load_to_lds(g0, g1, g2, g3, 0);
#endif
    __builtin_amdgcn_s_wait_tensorcnt(0);
  }
  __syncthreads();
#else
  for (int i = tid; i < DIM * DIM; i += 256) Wf32[i] = W[i];
  __syncthreads();
#endif

  // transpose + f16 convert inside LDS: Wlds[n*64+k] = W[k][n]
  for (int i = tid; i < DIM * DIM; i += 256) {
    int k = i >> 6, n = i & 63;
    Wlds[n * DIM + k] = (_Float16)Wf32[i];
  }
  __syncthreads();

  int wave = tid >> 5;
  int lane = tid & 31;
  int lm = lane & 15, half = lane >> 4;
  int nodeBase = blockIdx.x * 128 + wave * 16;

  // ---- A fragments (16x32 f16 tile per chunk); lane = M, half selects K window
  HFrag afrag[2];
  int row = nodeBase + lm;
  if (row >= N_NODES) row = N_NODES - 1;  // clamp: EXEC must stay all-ones for WMMA
  const float* xr = x + row * DIM;
#pragma unroll
  for (int kc = 0; kc < 2; ++kc) {
#pragma unroll
    for (int v = 0; v < 8; ++v) {
      int k = kc * 32 + (v >> 2) * 16 + half * 8 + (v & 3) * 2;
      afrag[kc].h[2 * v]     = (_Float16)xr[k];
      afrag[kc].h[2 * v + 1] = (_Float16)xr[k + 1];
    }
  }

  // ---- B fragments from LDS (32x16 f16): lane = N, half selects K window (same order as A)
  HFrag bfrag[4][2];
#pragma unroll
  for (int nt = 0; nt < 4; ++nt) {
    int n = nt * 16 + lm;
#pragma unroll
    for (int kc = 0; kc < 2; ++kc) {
#pragma unroll
      for (int v = 0; v < 8; ++v) {
        int k = kc * 32 + (v >> 2) * 16 + half * 8 + (v & 3) * 2;
        bfrag[nt][kc].u[v] = *(const unsigned int*)&Wlds[n * DIM + k];
      }
    }
  }

  // ---- 8 WMMAs: D = A*B + C
  v8f acc[4] = {};
#pragma unroll
  for (int kc = 0; kc < 2; ++kc)
#pragma unroll
    for (int nt = 0; nt < 4; ++nt)
      acc[nt] = __builtin_amdgcn_wmma_f32_16x16x32_f16(
          false, afrag[kc].v, false, bfrag[nt][kc].v, (short)0, acc[nt],
          false, false);

  // ---- fused attention scores + h store (C layout: lane = N, VGPR r = M)
  float aS[4], aD[4];
#pragma unroll
  for (int nt = 0; nt < 4; ++nt) {
    aS[nt] = a_src[nt * 16 + lm];
    aD[nt] = a_dst[nt * 16 + lm];
  }

  if (nodeBase + 16 <= N_NODES) {
    // full tile: wave-uniform path, no per-store exec masking
#pragma unroll
    for (int r = 0; r < 8; ++r) {
      int node = nodeBase + r + half * 8;
      float ps = 0.0f, pd = 0.0f;
#pragma unroll
      for (int nt = 0; nt < 4; ++nt) {
        float val = acc[nt][r];
        ps += val * aS[nt];
        pd += val * aD[nt];
        h[node * DIM + nt * 16 + lm] = val;
      }
#pragma unroll
      for (int off = 1; off < 16; off <<= 1) {
        ps += __shfl_xor(ps, off, 32);
        pd += __shfl_xor(pd, off, 32);
      }
      if (lm == 0) { ssrc[node] = ps; sdst[node] = pd; }
    }
  } else {
    // tail tile: guarded stores
#pragma unroll
    for (int r = 0; r < 8; ++r) {
      int node = nodeBase + r + half * 8;
      float ps = 0.0f, pd = 0.0f;
#pragma unroll
      for (int nt = 0; nt < 4; ++nt) {
        float val = acc[nt][r];
        ps += val * aS[nt];
        pd += val * aD[nt];
        if (node < N_NODES) h[node * DIM + nt * 16 + lm] = val;
      }
#pragma unroll
      for (int off = 1; off < 16; off <<= 1) {
        ps += __shfl_xor(ps, off, 32);
        pd += __shfl_xor(pd, off, 32);
      }
      if (lm == 0 && node < N_NODES) { ssrc[node] = ps; sdst[node] = pd; }
    }
  }
}

// ---------------------------------------------------------------------------
__device__ __forceinline__ void edge_ids(int e, const int* __restrict__ srcA,
                                         const int* __restrict__ dstA,
                                         int& s, int& d) {
  if (e < N_EDGES) { s = srcA[e]; d = dstA[e]; }
  else { s = e - N_EDGES; d = s; }  // appended self-loops
}

// alpha = leaky_relu(s_src[src] + s_dst[dst]); segment max via f32 atomic max
__global__ __launch_bounds__(256) void gat_edge_alpha(
    const int* __restrict__ srcA, const int* __restrict__ dstA,
    const float* __restrict__ ssrc, const float* __restrict__ sdst,
    float* __restrict__ alpha, float* amax) {
  int e = blockIdx.x * 256 + threadIdx.x;
  if (e >= TOT_E) return;
  int s, d;
  edge_ids(e, srcA, dstA, s, d);
  float a = ssrc[s] + sdst[d];
  a = (a > 0.0f) ? a : NEG_SLOPE * a;
  alpha[e] = a;
  atomicMax(amax + d, a);
}

// e = exp(alpha - amax[dst]); denom[dst] += e
__global__ __launch_bounds__(256) void gat_edge_exp(
    const int* __restrict__ srcA, const int* __restrict__ dstA,
    const float* __restrict__ alpha, const float* __restrict__ amax,
    float* __restrict__ eexp, float* denom) {
  int e = blockIdx.x * 256 + threadIdx.x;
  if (e >= TOT_E) return;
  int s, d;
  edge_ids(e, srcA, dstA, s, d);
  float ee = __expf(alpha[e] - amax[d]);
  eexp[e] = ee;
  atomicAdd(denom + d, ee);
}

// agg[dst] += (e/denom[dst]) * h[src]; wave-per-edge, coalesced 128B lines
__global__ __launch_bounds__(256) void gat_aggregate(
    const int* __restrict__ srcA, const int* __restrict__ dstA,
    const float* __restrict__ eexp, const float* __restrict__ denom,
    const float* __restrict__ h, float* agg) {
  int wave = threadIdx.x >> 5;
  int lane = threadIdx.x & 31;
  int e = blockIdx.x * 8 + wave;
  if (e >= TOT_E) return;
  int s, d;
  edge_ids(e, srcA, dstA, s, d);
  float coef = eexp[e] / denom[d];
  const float* hs = h + s * DIM;
  float* ad = agg + d * DIM;
  atomicAdd(ad + lane, coef * hs[lane]);
  atomicAdd(ad + lane + 32, coef * hs[lane + 32]);
}

// ---------------------------------------------------------------------------
__device__ __forceinline__ unsigned int hash_u32(unsigned int x) {
  x = x * 747796405u + 2891336453u;
  x = ((x >> ((x >> 28) + 4u)) ^ x) * 277803737u;
  return (x >> 22) ^ x;
}

// layer-1 epilogue: relu(agg + b) then deterministic dropout(p=0.3)
__global__ __launch_bounds__(256) void gat_finalize1(
    const float* __restrict__ agg, const float* __restrict__ b,
    float* __restrict__ x1) {
  int i = blockIdx.x * 256 + threadIdx.x;
  if (i >= N_NODES * DIM) return;
  float v = fmaxf(agg[i] + b[i & 63], 0.0f);
  unsigned int r = hash_u32((unsigned int)i ^ 0x9E3779B9u);
  bool keep = r < 0xB3333333u;  // P(keep) = 0.7
  x1[i] = keep ? v * (1.0f / KEEP_P) : 0.0f;
}

// layer-2 epilogue: relu(agg + b) -> output
__global__ __launch_bounds__(256) void gat_finalize2(
    const float* __restrict__ agg, const float* __restrict__ b,
    float* __restrict__ out) {
  int i = blockIdx.x * 256 + threadIdx.x;
  if (i >= N_NODES * DIM) return;
  out[i] = fmaxf(agg[i] + b[i & 63], 0.0f);
}

// ---------------------------------------------------------------------------
extern "C" void kernel_launch(void* const* d_in, const int* in_sizes, int n_in,
                              void* d_out, int out_size, void* d_ws,
                              size_t ws_size, hipStream_t stream) {
  const float* E   = (const float*)d_in[0];
  const int*   A   = (const int*)d_in[1];
  const float* W1  = (const float*)d_in[2];
  const float* as1 = (const float*)d_in[3];
  const float* ad1 = (const float*)d_in[4];
  const float* b1  = (const float*)d_in[5];
  const float* W2  = (const float*)d_in[6];
  const float* as2 = (const float*)d_in[7];
  const float* ad2 = (const float*)d_in[8];
  const float* b2  = (const float*)d_in[9];
  const int* srcA = A;
  const int* dstA = A + N_EDGES;

  float* w    = (float*)d_ws;
  float* H    = w; w += N_NODES * DIM;
  float* X1   = w; w += N_NODES * DIM;
  float* AGG  = w; w += N_NODES * DIM;
  float* SSRC = w; w += N_NODES;
  float* SDST = w; w += N_NODES;
  float* AMAX = w; w += N_NODES;
  float* DEN  = w; w += N_NODES;
  float* ALPH = w; w += TOT_E;
  float* EEXP = w; w += TOT_E;

  dim3 blk(256);
  int gInit = (N_NODES * DIM + 255) / 256;
  int gGemm = (N_NODES + 127) / 128;
  int gEdge = (TOT_E + 255) / 256;
  int gAggr = (TOT_E + 7) / 8;
  int gNode = (N_NODES * DIM + 255) / 256;

  // ---------------- layer 1 ----------------
  gat_init<<<gInit, blk, 0, stream>>>(AGG, AMAX, DEN);
  gat_gemm_attn<<<gGemm, blk, 0, stream>>>(E, W1, as1, ad1, H, SSRC, SDST);
  gat_edge_alpha<<<gEdge, blk, 0, stream>>>(srcA, dstA, SSRC, SDST, ALPH, AMAX);
  gat_edge_exp<<<gEdge, blk, 0, stream>>>(srcA, dstA, ALPH, AMAX, EEXP, DEN);
  gat_aggregate<<<gAggr, blk, 0, stream>>>(srcA, dstA, EEXP, DEN, H, AGG);
  gat_finalize1<<<gNode, blk, 0, stream>>>(AGG, b1, X1);

  // ---------------- layer 2 ----------------
  gat_init<<<gInit, blk, 0, stream>>>(AGG, AMAX, DEN);
  gat_gemm_attn<<<gGemm, blk, 0, stream>>>(X1, W2, as2, ad2, H, SSRC, SDST);
  gat_edge_alpha<<<gEdge, blk, 0, stream>>>(srcA, dstA, SSRC, SDST, ALPH, AMAX);
  gat_edge_exp<<<gEdge, blk, 0, stream>>>(srcA, dstA, ALPH, AMAX, EEXP, DEN);
  gat_aggregate<<<gAggr, blk, 0, stream>>>(srcA, dstA, EEXP, DEN, H, AGG);
  gat_finalize2<<<gNode, blk, 0, stream>>>(AGG, b2, (float*)d_out);
}
